// Reservoir_14104672600082
// MI455X (gfx1250) — matrix operand order repaired and checked by reading, not compiled, not run on hardware
//
#include <hip/hip_runtime.h>
#include <hip/hip_bf16.h>
#include <math.h>

#define T_STEPS 512
#define BATCH   64
#define DIN     256
#define HDIM    2048
#define KTOT    (DIN + HDIM)   // 2304
#define LDS_ROW 2312           // padded row stride (elements): +16B -> stride = 4 banks
#define NBLOCKS 32
#define ALPHA   0.9f

// sched_group_barrier masks
#define SG_VALU   0x002
#define SG_WMMA   0x008
#define SG_VMEM_R 0x020
#define SG_DS_R   0x100

typedef __attribute__((ext_vector_type(16))) __bf16       v16bf;
typedef __attribute__((ext_vector_type(8)))  float        v8f;
typedef __attribute__((ext_vector_type(4)))  float        v4f;
typedef __attribute__((ext_vector_type(4))) unsigned int  v4u;

union BF16x16 {
  v16bf  v;
  v4u    q[2];
  __bf16 e[16];
};

// 16-element bf16 fragment from two contiguous 16-byte chunks (global or LDS).
__device__ __forceinline__ v16bf load_bf16x16_pair(const __bf16* p0, const __bf16* p1) {
  BF16x16 r;
  r.q[0] = *(const v4u*)p0;
  r.q[1] = *(const v4u*)p1;
  return r.v;
}

// 16 fp32 (two 8-float runs) -> bf16 fragment (v_cvt_pk_bf16_f32).
__device__ __forceinline__ v16bf load_f32_as_bf16(const float* p0, const float* p1) {
  BF16x16 r;
  v4f a = *(const v4f*)(p0);
  v4f b = *(const v4f*)(p0 + 4);
  v4f c = *(const v4f*)(p1);
  v4f d = *(const v4f*)(p1 + 4);
#pragma unroll
  for (int i = 0; i < 4; ++i) {
    r.e[i]      = (__bf16)a[i];
    r.e[4 + i]  = (__bf16)b[i];
    r.e[8 + i]  = (__bf16)c[i];
    r.e[12 + i] = (__bf16)d[i];
  }
  return r.v;
}

__device__ __forceinline__ v8f wmma_bf16(v16bf a, v16bf b, v8f c) {
  return __builtin_amdgcn_wmma_f32_16x16x32_bf16(
      false, a, false, b, (short)0, c, false, false);
}

// fp32 -> bf16 conversion (weights + initial state), grid-stride.
__global__ void cvt_f32_to_bf16(const float* __restrict__ src,
                                __bf16* __restrict__ dst, int n) {
  int i = blockIdx.x * blockDim.x + threadIdx.x;
  int stride = gridDim.x * blockDim.x;
  for (; i < n; i += stride) dst[i] = (__bf16)src[i];
}

__global__ void zero_u32(unsigned* p) { *p = 0u; }

// Persistent reservoir kernel.
// 32 blocks x 256 threads. Block b owns output columns [b*64, b*64+64).
// Its full weight strip [W_in | W_hat] (64 rows x 2304 bf16, padded) lives in
// LDS for all 512 timesteps. Per step: 8 waves each compute a 16x32 tile
// (1 M-tile x 2 N-tiles) with bf16 WMMA. A fragments (global, L2) are
// pipelined at distance 2; B fragments (LDS) at distance 1; scheduling
// groups pin the shape: [vmem reads][ds reads][wmma] per iteration.
// Steps are separated by a device-wide atomic barrier (release/acquire).
__global__ __launch_bounds__(256) void reservoir_persist(
    const float*  __restrict__ inp,    // [T][BATCH][DIN] fp32
    const float*  __restrict__ init0,  // [BATCH][HDIM]   fp32
    const __bf16* __restrict__ Winb,   // [HDIM][DIN]     bf16
    const __bf16* __restrict__ Whatb,  // [HDIM][HDIM]    bf16
    __bf16*       __restrict__ sbuf0,  // [BATCH][HDIM]   bf16 state ping
    __bf16*       __restrict__ sbuf1,  // [BATCH][HDIM]   bf16 state pong
    float*        __restrict__ out,    // [T][BATCH][HDIM] fp32
    unsigned*     __restrict__ bar)    // device-wide barrier counter
{
  extern __shared__ __bf16 lds[];      // [64][LDS_ROW]
  const int tid   = threadIdx.x;
  const int lane  = tid & 31;
  const int wavei = tid >> 5;          // 0..7
  const int n0    = blockIdx.x * 64;   // strip origin in H
  const int m0    = (wavei & 3) * 16;  // batch-tile origin
  const int noff  = (wavei >> 2) * 32; // half-strip origin
  const int r     = lane & 15;
  const int hi    = lane >> 4;         // 0/1 half-wave

  // ---- stage the weight strip into LDS once (b128 global -> b128 ds) ----
  for (int idx = tid; idx < 64 * (DIN / 8); idx += 256) {
    const int rr = idx >> 5, c = (idx & 31) * 8;
    *(v4u*)(lds + rr * LDS_ROW + c) =
        *(const v4u*)(Winb + (size_t)(n0 + rr) * DIN + c);
  }
  for (int idx = tid; idx < 64 * (HDIM / 8); idx += 256) {
    const int rr = idx >> 8, c = (idx & 255) * 8;
    *(v4u*)(lds + rr * LDS_ROW + DIN + c) =
        *(const v4u*)(Whatb + (size_t)(n0 + rr) * HDIM + c);
  }
  __syncthreads();

  // Per-lane LDS base pointers for the two B tiles of this wave.
  const __bf16* ldsb0 = lds + (size_t)(noff +  0 + r) * LDS_ROW + hi * 16;
  const __bf16* ldsb1 = lds + (size_t)(noff + 16 + r) * LDS_ROW + hi * 16;

  const float al = ALPHA, om = 1.0f - ALPHA;

  for (int t = 0; t < T_STEPS; ++t) {
    const __bf16* s_in  = (t & 1) ? sbuf1 : sbuf0;
    __bf16*       s_out = (t & 1) ? sbuf0 : sbuf1;
    const float*  prev  = (t == 0) ? init0 : out + (size_t)(t - 1) * BATCH * HDIM;
    const float*  x_t   = inp + (size_t)t * BATCH * DIN;

    // warm the epilogue's prev-state line early
    __builtin_prefetch(prev + (size_t)(m0 + hi * 8) * HDIM + n0 + noff + r, 0, 1);

    v8f acc[2] = {v8f{0.f}, v8f{0.f}};

    // ---- phase 1: x_t @ W_in^T  (K = 0..255, A fp32 -> bf16 on the fly) ----
    {
      const float* arow = x_t + (size_t)(m0 + r) * DIN + hi * 16;
      v16bf a  = load_f32_as_bf16(arow, arow + 8);
      v16bf f0 = load_bf16x16_pair(ldsb0, ldsb0 + 8);
      v16bf f1 = load_bf16x16_pair(ldsb1, ldsb1 + 8);
#pragma unroll
      for (int k = 0; k < DIN - 32; k += 32) {
        v16bf an = load_f32_as_bf16(arow + k + 32, arow + k + 40);
        v16bf g0 = load_bf16x16_pair(ldsb0 + k + 32, ldsb0 + k + 40);
        v16bf g1 = load_bf16x16_pair(ldsb1 + k + 32, ldsb1 + k + 40);
        acc[0] = wmma_bf16(a, f0, acc[0]);
        acc[1] = wmma_bf16(a, f1, acc[1]);
        a = an; f0 = g0; f1 = g1;
        // pin schedule: next loads first, cvts, then current WMMAs
        __builtin_amdgcn_sched_group_barrier(SG_VMEM_R, 4, 0);
        __builtin_amdgcn_sched_group_barrier(SG_DS_R,   4, 0);
        __builtin_amdgcn_sched_group_barrier(SG_VALU,   8, 0);
        __builtin_amdgcn_sched_group_barrier(SG_WMMA,   2, 0);
      }
      acc[0] = wmma_bf16(a, f0, acc[0]);
      acc[1] = wmma_bf16(a, f1, acc[1]);
    }

    // ---- phase 2: state @ W_hat^T  (K = 0..2047, A bf16, B from LDS) ----
    {
      const __bf16* arow = s_in + (size_t)(m0 + r) * HDIM + hi * 16;
      v16bf a0 = load_bf16x16_pair(arow, arow + 8);
      v16bf a1 = load_bf16x16_pair(arow + 32, arow + 40);
      v16bf f0 = load_bf16x16_pair(ldsb0 + DIN, ldsb0 + DIN + 8);
      v16bf f1 = load_bf16x16_pair(ldsb1 + DIN, ldsb1 + DIN + 8);
#pragma unroll 2
      for (int k = 0; k < HDIM - 64; k += 32) {
        v16bf an = load_bf16x16_pair(arow + k + 64, arow + k + 72);
        v16bf g0 = load_bf16x16_pair(ldsb0 + DIN + k + 32, ldsb0 + DIN + k + 40);
        v16bf g1 = load_bf16x16_pair(ldsb1 + DIN + k + 32, ldsb1 + DIN + k + 40);
        acc[0] = wmma_bf16(a0, f0, acc[0]);
        acc[1] = wmma_bf16(a0, f1, acc[1]);
        a0 = a1; a1 = an; f0 = g0; f1 = g1;
        // pin schedule: [A loads][B ds loads][2 WMMA]
        __builtin_amdgcn_sched_group_barrier(SG_VMEM_R, 2, 0);
        __builtin_amdgcn_sched_group_barrier(SG_DS_R,   4, 0);
        __builtin_amdgcn_sched_group_barrier(SG_WMMA,   2, 0);
      }
      {  // tail: chunks HDIM-64 and HDIM-32
        const int k = HDIM - 64;
        v16bf g0 = load_bf16x16_pair(ldsb0 + DIN + k + 32, ldsb0 + DIN + k + 40);
        v16bf g1 = load_bf16x16_pair(ldsb1 + DIN + k + 32, ldsb1 + DIN + k + 40);
        acc[0] = wmma_bf16(a0, f0, acc[0]);
        acc[1] = wmma_bf16(a0, f1, acc[1]);
        acc[0] = wmma_bf16(a1, g0, acc[0]);
        acc[1] = wmma_bf16(a1, g1, acc[1]);
      }
    }

    // ---- epilogue: tanh + leaky blend ----
#pragma unroll
    for (int i = 0; i < 8; ++i) {
      const int m = m0 + hi * 8 + i;
#pragma unroll
      for (int j = 0; j < 2; ++j) {
        const int n = n0 + noff + j * 16 + r;
        const size_t idx = (size_t)m * HDIM + n;
        const float ns = om * prev[idx] + al * tanhf(acc[j][i]);
        out[(size_t)t * BATCH * HDIM + idx] = ns;
        s_out[idx] = (__bf16)ns;
      }
    }

    // ---- device-wide barrier between timesteps ----
    __threadfence();   // make this wave's stores device-visible
    __syncthreads();   // all waves in block arrived (stores fenced)
    if (tid == 0)
      __hip_atomic_fetch_add(bar, 1u, __ATOMIC_RELEASE, __HIP_MEMORY_SCOPE_AGENT);
    const unsigned target = (unsigned)NBLOCKS * (unsigned)(t + 1);
    if (lane == 0) {
      while (__hip_atomic_load(bar, __ATOMIC_ACQUIRE, __HIP_MEMORY_SCOPE_AGENT) < target) {
        __builtin_amdgcn_s_sleep(1);
      }
    }
    // wave reconverges here; EXEC all-1s again before next step's WMMAs
  }
}

extern "C" void kernel_launch(void* const* d_in, const int* in_sizes, int n_in,
                              void* d_out, int out_size, void* d_ws, size_t ws_size,
                              hipStream_t stream) {
  (void)in_sizes; (void)n_in; (void)out_size; (void)ws_size;
  const float* inp  = (const float*)d_in[0];  // [T, B, DIN]
  const float* init = (const float*)d_in[1];  // [B, H]
  const float* Win  = (const float*)d_in[2];  // [H, DIN]
  const float* What = (const float*)d_in[3];  // [H, H]
  float*       out  = (float*)d_out;          // [T, B, H]

  // Workspace: bf16 W_in | bf16 W_hat | state ping | state pong | barrier
  __bf16* Winb  = (__bf16*)d_ws;
  __bf16* Whatb = Winb  + (size_t)HDIM * DIN;
  __bf16* sbuf0 = Whatb + (size_t)HDIM * HDIM;
  __bf16* sbuf1 = sbuf0 + (size_t)BATCH * HDIM;
  unsigned* bar = (unsigned*)((char*)d_ws + (10u << 20));  // past ~9.5MB of buffers

  const int lds_bytes = 64 * LDS_ROW * 2;  // 295,936 B < 320 KB WGP LDS
  (void)hipFuncSetAttribute((const void*)reservoir_persist,
                            hipFuncAttributeMaxDynamicSharedMemorySize, lds_bytes);

  // Per-launch setup.
  {
    int n = HDIM * DIN;
    cvt_f32_to_bf16<<<(n + 255) / 256, 256, 0, stream>>>(Win, Winb, n);
    n = HDIM * HDIM;
    cvt_f32_to_bf16<<<(n + 255) / 256, 256, 0, stream>>>(What, Whatb, n);
    n = BATCH * HDIM;
    cvt_f32_to_bf16<<<(n + 255) / 256, 256, 0, stream>>>(init, sbuf0, n);
    zero_u32<<<1, 1, 0, stream>>>(bar);
  }

  // One persistent launch runs all 512 timesteps.
  reservoir_persist<<<NBLOCKS, 256, lds_bytes, stream>>>(
      inp, init, Winb, Whatb, sbuf0, sbuf1, out, bar);
}
